// ViscoplasticMaterialModel_40647570489683
// MI455X (gfx1250) — matrix operand order repaired and checked by reading, not compiled, not run on hardware
//
#include <hip/hip_runtime.h>
#include <math.h>

typedef __attribute__((ext_vector_type(16))) _Float16 v16h;
typedef __attribute__((ext_vector_type(8)))  _Float16 v8h;
typedef __attribute__((ext_vector_type(8)))  float    v8f;

#define B_TOT   8192
#define T_STEPS 64
#define L_PTS   128
#define N_MODES 16
#define W_DIM   32
#define WF_DIM  64
#define O_DIM   8
#define BT      8      // samples per block
#define NTHREADS 256   // 8 waves (wave32)
#define DT_STEP 0.01f

__device__ __forceinline__ int lane_id() { return (int)(threadIdx.x & 31u); }

__device__ __forceinline__ v8f wmma16(v16h a, v16h b, v8f c) {
  // v_wmma_f32_16x16x32_f16 : D = A(16x32) * B(32x16) + C
  return __builtin_amdgcn_wmma_f32_16x16x32_f16(false, a, false, b, (short)0, c,
                                                false, false);
}

__device__ __forceinline__ v16h combine8(v8h lo, v8h hi) {
  v16h r;
#pragma unroll
  for (int j = 0; j < 8; ++j) { r[j] = lo[j]; r[8 + j] = hi[j]; }
  return r;
}

// A fragment (16x32 f16) from row-major [M][K] storage; ld (halves) multiple of 8.
// Lane n<16: M=n, K = 0..15 ; lane>=16: M=n-16, K = 16..31
__device__ __forceinline__ v16h load_a16(const _Float16* base, int ld) {
  const int lane = lane_id();
  const int row  = lane & 15;
  const int kb   = (lane >> 4) << 4;
  const _Float16* p = base + row * ld + kb;
  return combine8(*(const v8h*)(p), *(const v8h*)(p + 8));
}

// B fragment (32x16 f16) from N-major storage bT[N][K]; ldk (halves) multiple of 8.
// Lane n<16: N=n, K = 0..15 ; lane>=16: N=n-16, K = 16..31
__device__ __forceinline__ v16h load_bT(const _Float16* baseT, int ldk) {
  const int lane = lane_id();
  const int col  = lane & 15;
  const int kb   = (lane >> 4) << 4;
  const _Float16* p = baseT + col * ldk + kb;
  return combine8(*(const v8h*)(p), *(const v8h*)(p + 8));
}

// A fragment for the mode-mix GEMM from ftT[m][re|im][s][i] (i contiguous).
// second=false: [ft_re ; ft_im]   second=true: [-ft_im ; ft_re]
__device__ __forceinline__ v16h load_ft_frag(const _Float16* ftT, int m, bool second) {
  const int lane = lane_id();
  const int row  = lane & 15;
  const int kb   = (lane >> 4) << 4;
  const int s    = row & 7;
  const bool hi  = row >= 8;
  const int ri   = second ? (hi ? 0 : 1) : (hi ? 1 : 0);
  const bool neg = second && !hi;
  const _Float16* p = ftT + ((m * 2 + ri) * BT + s) * W_DIM + kb;
  v8h lo = *(const v8h*)(p);
  v8h hv = *(const v8h*)(p + 8);
  v16h a;
#pragma unroll
  for (int j = 0; j < 8; ++j) {
    a[j]     = neg ? (_Float16)(-(float)lo[j]) : lo[j];
    a[8 + j] = neg ? (_Float16)(-(float)hv[j]) : hv[j];
  }
  return a;
}

// B fragment from global spectral weight (W,W,MODES) fp32: column slice o0.., mode m.
__device__ __forceinline__ v16h load_spec_frag(const float* __restrict__ spec,
                                               int m, int o0) {
  const int lane = lane_id();
  const int col  = lane & 15;
  const int kb   = (lane >> 4) << 4;
  v16h b;
#pragma unroll
  for (int j = 0; j < 16; ++j)
    b[j] = (_Float16)spec[((kb + j) * W_DIM + o0 + col) * N_MODES + m];
  return b;
}

__device__ __forceinline__ float gelu_tanh(float x) {
  const float k0 = 0.7978845608028654f; // sqrt(2/pi)
  float x3 = x * x * x;
  return 0.5f * x * (1.0f + tanhf(k0 * (x + 0.044715f * x3)));
}

__device__ __forceinline__ float softplus_f(float x) {
  return (x > 20.0f) ? x : log1pf(expf(x));
}

// ---------------------------------------------------------------------------
// Kernel 1: FNF1d feature net.  Block = 8 samples resident in LDS, 8 waves.
// All heavy linear algebra on v_wmma_f32_16x16x32_f16; every LDS fragment
// load is a contiguous 16B chunk (ds_load_b128); some fragment stores are 2B scatters.
// ---------------------------------------------------------------------------
__global__ __launch_bounds__(NTHREADS)
void fnf_kernel(const float* __restrict__ Yg, const float* __restrict__ ng,
                const float* __restrict__ edg,
                const float* __restrict__ lift, const float* __restrict__ proj1,
                const float* __restrict__ proj2,
                const float* __restrict__ pw0, const float* __restrict__ pw1,
                const float* __restrict__ pw2,
                const float* __restrict__ s0re, const float* __restrict__ s0im,
                const float* __restrict__ s1re, const float* __restrict__ s1im,
                const float* __restrict__ s2re, const float* __restrict__ s2im,
                float* __restrict__ m2_out) {
  __shared__ alignas(16) _Float16 sh_hwl[BT * W_DIM * L_PTS];  // 64 KB h[s][w][l]
  __shared__ alignas(16) _Float16 sh_hlw[BT * L_PTS * W_DIM];  // 64 KB h[s][l][w]
  __shared__ alignas(16) _Float16 sh_ftT[N_MODES * 2 * BT * W_DIM]; // 16 KB ftT[m][ri][s][i]
  __shared__ alignas(16) _Float16 sh_out[BT * W_DIM * 32];     // 16 KB out[s][o][re16|im16]
  __shared__ alignas(16) _Float16 sh_dftT[32 * L_PTS];         //  8 KB [mcol][l]
  __shared__ alignas(16) _Float16 sh_idftT[L_PTS * 32];        //  8 KB [l][kcat]
  __shared__ alignas(16) _Float16 sh_pwt[W_DIM * W_DIM];       //  2 KB pw^T[v][w]
  __shared__ float sh_pool[BT * W_DIM];
  __shared__ float sh_f1[BT * WF_DIM];

  const int tid  = threadIdx.x;
  const int wave = tid >> 5;
  const int b0   = blockIdx.x * BT;
  const float TWO_PI = 6.283185307179586f;

  // L2 prefetch of heavily-reused spectral weights (global_prefetch_b8)
  __builtin_prefetch(s0re + (tid << 4), 0, 1);
  __builtin_prefetch(s0im + (tid << 4), 0, 1);

  // --- twiddle tables (truncated DFT as matmul), transposed for B-frags ---
  for (int idx = tid; idx < 32 * L_PTS; idx += NTHREADS) {
    int mcol = idx >> 7, l = idx & 127;             // dftT[mcol][l]
    int m = mcol & 15;
    float ang = TWO_PI * (float)(l * m) / (float)L_PTS;
    sh_dftT[idx] = (_Float16)((mcol < 16) ? cosf(ang) : -sinf(ang));
  }
  for (int idx = tid; idx < L_PTS * 32; idx += NTHREADS) {
    int l = idx >> 5, kc = idx & 31;                // idftT[l][kcat]
    int m = kc & 15;
    float ang = TWO_PI * (float)(m * l) / (float)L_PTS;
    float v;
    if (kc < 16) v = ((m == 0) ? 1.0f : 2.0f) * cosf(ang) * (1.0f / L_PTS);
    else         v = (m == 0) ? 0.0f : (-2.0f * sinf(ang) * (1.0f / L_PTS));
    sh_idftT[idx] = (_Float16)v;
  }
  // --- lift: h[s][w][l] = Y*lift[0][w] + n*lift[1][w] + edot*lift[2][w] ---
  for (int idx = tid; idx < BT * L_PTS; idx += NTHREADS) {
    int s = idx >> 7, l = idx & 127;
    int g = (b0 + s) * L_PTS + l;
    float y = Yg[g], nn = ng[g], e0 = edg[g];
#pragma unroll
    for (int wg = 0; wg < 4; ++wg) {
      v8h pk;
#pragma unroll
      for (int j = 0; j < 8; ++j) {
        int w = wg * 8 + j;
        float hv = y * lift[w] + nn * lift[W_DIM + w] + e0 * lift[2 * W_DIM + w];
        pk[j] = (_Float16)hv;
        sh_hwl[(s * W_DIM + w) * L_PTS + l] = pk[j];
      }
      *(v8h*)(sh_hlw + (s * L_PTS + l) * W_DIM + wg * 8) = pk;
    }
  }
  __syncthreads();

  const float* pws[3] = {pw0, pw1, pw2};
  const float* sre[3] = {s0re, s1re, s2re};
  const float* sim[3] = {s0im, s1im, s2im};

  for (int k = 0; k < 3; ++k) {
    // stage pw^T for this layer
    for (int idx = tid; idx < W_DIM * W_DIM; idx += NTHREADS) {
      int v = idx >> 5, w = idx & 31;
      sh_pwt[v * W_DIM + w] = (_Float16)pws[k][w * W_DIM + v];
    }
    __syncthreads();

    // GEMM A: ft = h x DFT   rows=(s,i), K=128, N=32(re|im).  Wave owns sample.
    {
      const int s = wave;
      for (int it = 0; it < 2; ++it) {
        for (int nt = 0; nt < 2; ++nt) {
          v8f acc = {};
#pragma unroll
          for (int kk = 0; kk < 4; ++kk) {
            v16h a = load_a16(sh_hwl + (s * W_DIM + it * 16) * L_PTS + kk * 32, L_PTS);
            v16h b = load_bT(sh_dftT + (nt * 16) * L_PTS + kk * 32, L_PTS);
            acc = wmma16(a, b, acc);
          }
          // store contiguous into ftT[m][ri=nt][s][i]
          const int lane = lane_id();
          const int m  = lane & 15;
          const int i0 = it * 16 + ((lane >> 4) << 3);
          v8h pk;
#pragma unroll
          for (int r = 0; r < 8; ++r) pk[r] = (_Float16)acc[r];
          *(v8h*)(sh_ftT + ((m * 2 + nt) * BT + s) * W_DIM + i0) = pk;
        }
      }
    }
    __syncthreads();

    // GEMM B: complex mode mix. C = [ftre;ftim]xWre + [-ftim;ftre]xWim
    {
      for (int j = 0; j < 2; ++j) {
        const int m = wave * 2 + j;
        for (int ot = 0; ot < 2; ++ot) {
          v8f acc = {};
          acc = wmma16(load_ft_frag(sh_ftT, m, false),
                       load_spec_frag(sre[k], m, ot * 16), acc);
          acc = wmma16(load_ft_frag(sh_ftT, m, true),
                       load_spec_frag(sim[k], m, ot * 16), acc);
          const int lane = lane_id();
          const bool hi  = lane >= 16;            // C rows 8..15 => imag part
          const int o    = ot * 16 + (lane & 15);
          const int moff = hi ? (16 + m) : m;
#pragma unroll
          for (int r = 0; r < 8; ++r)           // scatter store (no load-wait cost)
            sh_out[(r * W_DIM + o) * 32 + moff] = (_Float16)acc[r];
        }
      }
    }
    __syncthreads();

    // GEMM C+D fused: h = [gelu]( IDFT(out) + pw^T x h ), in place (wave==sample)
    {
      const int s = wave;
      for (int lt = 0; lt < 8; ++lt) {
        v8f acc0 = {}, acc1 = {};
        {
          v16h b = load_bT(sh_idftT + (lt * 16) * 32, 32);
          acc0 = wmma16(load_a16(sh_out + (s * W_DIM + 0) * 32, 32), b, acc0);
          acc1 = wmma16(load_a16(sh_out + (s * W_DIM + 16) * 32, 32), b, acc1);
        }
        {
          v16h b = load_bT(sh_hlw + (s * L_PTS + lt * 16) * W_DIM, W_DIM);
          acc0 = wmma16(load_a16(sh_pwt + 0, W_DIM), b, acc0);
          acc1 = wmma16(load_a16(sh_pwt + 16 * W_DIM, W_DIM), b, acc1);
        }
        const int lane = lane_id();
        const int rr = (lane >> 4) << 3;
        const int cc = lt * 16 + (lane & 15);     // l index
        v8h p0, p1;
#pragma unroll
        for (int r = 0; r < 8; ++r) {
          float v0 = acc0[r], v1 = acc1[r];
          if (k < 2) { v0 = gelu_tanh(v0); v1 = gelu_tanh(v1); }
          p0[r] = (_Float16)v0;
          p1[r] = (_Float16)v1;
          sh_hwl[(s * W_DIM + (rr + r)) * L_PTS + cc]      = p0[r];
          sh_hwl[(s * W_DIM + (16 + rr + r)) * L_PTS + cc] = p1[r];
        }
        *(v8h*)(sh_hlw + (s * L_PTS + cc) * W_DIM + rr)      = p0;
        *(v8h*)(sh_hlw + (s * L_PTS + cc) * W_DIM + 16 + rr) = p1;
      }
    }
    __syncthreads();
  }

  // --- mean pool over L, then proj1(gelu) / proj2 ---
  {
    const int s = tid >> 5, w = tid & 31;
    float acc = 0.f;
    for (int l = 0; l < L_PTS; l += 8) {
      v8h pk = *(const v8h*)(sh_hwl + (s * W_DIM + w) * L_PTS + l);
#pragma unroll
      for (int j = 0; j < 8; ++j) acc += (float)pk[j];
    }
    sh_pool[s * W_DIM + w] = acc * (1.0f / L_PTS);
  }
  __syncthreads();
  for (int idx = tid; idx < BT * WF_DIM; idx += NTHREADS) {
    int s = idx >> 6, j = idx & 63;
    float acc = 0.f;
    for (int w = 0; w < W_DIM; ++w) acc += sh_pool[s * W_DIM + w] * proj1[w * WF_DIM + j];
    sh_f1[idx] = gelu_tanh(acc);
  }
  __syncthreads();
  if (tid < BT * O_DIM) {
    int s = tid >> 3, d = tid & 7;
    float acc = 0.f;
    for (int j = 0; j < WF_DIM; ++j) acc += sh_f1[s * WF_DIM + j] * proj2[j * O_DIM + d];
    m2_out[(b0 + s) * O_DIM + d] = acc;
  }
}

// ---------------------------------------------------------------------------
// Kernel 2: PICNN Newton time loop.  One lane per sample; D'(q) collapsed to
// lin + sum_j g1_j*sigmoid(b_j q + c_j), D''(q) = sum_j g2_j*sig*(1-sig).
// ---------------------------------------------------------------------------
__global__ __launch_bounds__(256)
void solve_kernel(const float* __restrict__ e, const float* __restrict__ E,
                  const float* __restrict__ m2g, const float* __restrict__ energy,
                  const float* __restrict__ Wuy, const float* __restrict__ Wuz,
                  const float* __restrict__ Wxu, const float* __restrict__ Wxy,
                  const float* __restrict__ Wxz, const float* __restrict__ Wy0,
                  const float* __restrict__ Wyout, const float* __restrict__ Wzout,
                  float* __restrict__ out_stress, float* __restrict__ out_xi) {
  const int b = blockIdx.x * blockDim.x + threadIdx.x;
  if (b >= B_TOT) return;

  // harmonic mean of E
  float inv = 0.f;
  for (int l = 0; l < L_PTS; ++l) inv += 1.0f / E[b * L_PTS + l];
  const float m1 = (float)L_PTS / inv;

  float x[O_DIM];
#pragma unroll
  for (int d = 0; d < O_DIM; ++d) x[d] = m2g[b * O_DIM + d];

  float u[16];
#pragma unroll
  for (int kk = 0; kk < 16; ++kk) {
    float t = 0.f;
#pragma unroll
    for (int d = 0; d < O_DIM; ++d) t += x[d] * Wxu[d * 16 + kk];
    u[kk] = softplus_f(t);
  }
  float a = 0.f;
#pragma unroll
  for (int d = 0; d < O_DIM; ++d) a += x[d] * Wxy[d];
  float lin = 0.f;
#pragma unroll
  for (int kk = 0; kk < 16; ++kk) lin += u[kk] * Wuy[kk];
  lin *= Wyout[0];

  float cj[16], bj[16], g1[16], g2[16];
#pragma unroll
  for (int j = 0; j < 16; ++j) {
    float cc = 0.f;
#pragma unroll
    for (int d = 0; d < O_DIM; ++d) cc += x[d] * Wxz[d * 16 + j];
    cj[j] = cc;
    float sz = 0.f;
#pragma unroll
    for (int kk = 0; kk < 16; ++kk) sz += u[kk] * Wuz[kk * 16 + j];
    float sj  = softplus_f(sz);
    float wzj = softplus_f(Wzout[j]);
    bj[j] = a * Wy0[j];
    g1[j] = sj * wzj * bj[j];
    g2[j] = g1[j] * bj[j];
  }

  const float twopm  = 2.0f * energy[0] * m1;
  const float inv_dt = 1.0f / DT_STEP;
  float xi_prev = 0.f, xi_prev2 = 0.f;
  out_xi[b * T_STEPS] = 0.f;

  for (int t = 0; t < T_STEPS; ++t) {
    const float uu = e[b * T_STEPS + t];
    float xi = (t > 2) ? (2.0f * xi_prev - xi_prev2) : xi_prev;
    float s_eq = 0.f;
#pragma unroll 1
    for (int it = 0; it < 10; ++it) {
      const float s_new = twopm * (uu - xi);
      const float xd = (xi - xi_prev) * inv_dt;
      float Dd = lin, Ddd = 0.f;
#pragma unroll
      for (int j = 0; j < 16; ++j) {
        const float z  = bj[j] * xd + cj[j];
        const float sg = 1.0f / (1.0f + expf(-z));
        Dd  += g1[j] * sg;
        Ddd += g2[j] * sg * (1.0f - sg);
      }
      const float grad = Dd - s_new;                 // w_d + D_d
      xi = xi - grad / (twopm + Ddd * inv_dt);       // Newton step
      s_eq = s_new;
    }
    out_stress[b * T_STEPS + t] = s_eq;
    if (t < T_STEPS - 1) out_xi[b * T_STEPS + t + 1] = xi;
    xi_prev2 = xi_prev;
    xi_prev  = xi;
  }
}

// ---------------------------------------------------------------------------
// Input order = jax pytree flatten of setup_inputs():
//  0:e 1:E 2:Y 3:n 4:edot_0 5:energy_param
//  6:lift 7:proj1 8:proj2 9:pw0 10:pw1 11:pw2
//  12:spec0_im 13:spec0_re 14:spec1_im 15:spec1_re 16:spec2_im 17:spec2_re
//  18:Wuout 19:Wuy 20:Wuz 21:Wxu 22:Wxy 23:Wxz 24:Wy0 25:Wyout 26:Wzout
// ---------------------------------------------------------------------------
extern "C" void kernel_launch(void* const* d_in, const int* in_sizes, int n_in,
                              void* d_out, int out_size, void* d_ws, size_t ws_size,
                              hipStream_t stream) {
  (void)in_sizes; (void)n_in; (void)out_size; (void)ws_size;
  const float* e     = (const float*)d_in[0];
  const float* E     = (const float*)d_in[1];
  const float* Y     = (const float*)d_in[2];
  const float* nn    = (const float*)d_in[3];
  const float* ed    = (const float*)d_in[4];
  const float* energ = (const float*)d_in[5];
  const float* lift  = (const float*)d_in[6];
  const float* proj1 = (const float*)d_in[7];
  const float* proj2 = (const float*)d_in[8];
  const float* pw0   = (const float*)d_in[9];
  const float* pw1   = (const float*)d_in[10];
  const float* pw2   = (const float*)d_in[11];
  const float* s0im  = (const float*)d_in[12];
  const float* s0re  = (const float*)d_in[13];
  const float* s1im  = (const float*)d_in[14];
  const float* s1re  = (const float*)d_in[15];
  const float* s2im  = (const float*)d_in[16];
  const float* s2re  = (const float*)d_in[17];
  const float* Wuy   = (const float*)d_in[19];
  const float* Wuz   = (const float*)d_in[20];
  const float* Wxu   = (const float*)d_in[21];
  const float* Wxy   = (const float*)d_in[22];
  const float* Wxz   = (const float*)d_in[23];
  const float* Wy0   = (const float*)d_in[24];
  const float* Wyout = (const float*)d_in[25];
  const float* Wzout = (const float*)d_in[26];

  float* m2g = (float*)d_ws;                 // B x 8 features
  float* out = (float*)d_out;                // [stress | xi]

  fnf_kernel<<<B_TOT / BT, NTHREADS, 0, stream>>>(
      Y, nn, ed, lift, proj1, proj2, pw0, pw1, pw2,
      s0re, s0im, s1re, s1im, s2re, s2im, m2g);

  solve_kernel<<<B_TOT / 256, 256, 0, stream>>>(
      e, E, m2g, energ, Wuy, Wuz, Wxu, Wxy, Wxz, Wy0, Wyout, Wzout,
      out, out + B_TOT * T_STEPS);
}
